// EdgeBiasAttention_45200235823669
// MI455X (gfx1250) — matrix operand order repaired and checked by reading, not compiled, not run on hardware
//
#include <hip/hip_runtime.h>
#include <hip/hip_bf16.h>

typedef __attribute__((ext_vector_type(16))) __bf16 v16bf;
typedef __attribute__((ext_vector_type(8)))  float  v8f;

#define C_DIM 64
#define BATCH 2
#define HID   16

// ---------------------------------------------------------------------------
// Kernel 1: per-edge bias MLP  (2 -> 16 -> 1, relu)
// ---------------------------------------------------------------------------
__global__ __launch_bounds__(256) void k_bias(
    const float* __restrict__ efeat, const float* __restrict__ W1,
    const float* __restrict__ b1, const float* __restrict__ W2,
    const float* __restrict__ b2, float* __restrict__ bias, int E)
{
    int e = blockIdx.x * blockDim.x + threadIdx.x;
    if (e >= E) return;
    float e0 = efeat[(size_t)e * 2 + 0];
    float e1 = efeat[(size_t)e * 2 + 1];
    float acc = b2[0];
#pragma unroll
    for (int j = 0; j < HID; ++j) {
        float h = fmaf(e0, W1[j], fmaf(e1, W1[HID + j], b1[j]));
        h = h > 0.0f ? h : 0.0f;
        acc = fmaf(h, W2[j], acc);
    }
    bias[e] = acc;
}

// ---------------------------------------------------------------------------
// Kernel 2: per-edge logits via WMMA (16 edges per wave, diagonal of
// Q[dst-rows] x K[src-rows]^T), compensated bf16 split for ~f32 accuracy,
// plus atomic segment-max with a monotone uint key.
// ---------------------------------------------------------------------------
__global__ __launch_bounds__(256) void k_logits(
    const float* __restrict__ Q, const float* __restrict__ K,
    const int* __restrict__ src, const int* __restrict__ dst,
    const float* __restrict__ bias,
    float* __restrict__ logit, unsigned* __restrict__ smax,
    int N, int E)
{
    int tid  = blockIdx.x * blockDim.x + threadIdx.x;
    int wave = tid >> 5;                 // wave-uniform
    int lane = threadIdx.x & 31;
    int batch = wave & 1;
    int group = wave >> 1;
    int nGroups = (E + 15) >> 4;
    if (group >= nGroups) return;        // uniform exit: EXEC stays all-1 for WMMA

    int eBase = group * 16;
    int i = lane & 15;
    int e = eBase + i; if (e >= E) e = E - 1;   // clamp (writes predicated later)
    int drow = dst[e];
    int srow = src[e];
    const float* qrow = Q + ((size_t)batch * N + drow) * C_DIM;
    const float* krow = K + ((size_t)batch * N + srow) * C_DIM;

    const bool hiHalf = lane >= 16;
    const int  klo  = hiHalf ? 8  : 0;   // A-fragment K base within 32-chunk
    const int  koff = hiHalf ? 16 : 0;   // unused; B fragment also uses klo

    v8f acc = {0.f, 0.f, 0.f, 0.f, 0.f, 0.f, 0.f, 0.f};

#pragma unroll
    for (int chunk = 0; chunk < C_DIM; chunk += 32) {
        // --- A fragment (16x32 bf16): elems 0..7 = q[chunk+klo+j], 8..15 = q[chunk+klo+16+j]
        v16bf ah, al;
#pragma unroll
        for (int j = 0; j < 8; ++j) {
            float f0 = qrow[chunk + klo + j];
            float f1 = qrow[chunk + klo + 16 + j];
            __bf16 h0 = (__bf16)f0;
            __bf16 h1 = (__bf16)f1;
            ah[j]     = h0;  al[j]     = (__bf16)(f0 - (float)h0);
            ah[8 + j] = h1;  al[8 + j] = (__bf16)(f1 - (float)h1);
        }
        // --- B fragment (32x16 bf16): elems 0..7 = k[chunk+klo+j], 8..15 = k[chunk+klo+16+j]
        v16bf bh, bl;
#pragma unroll
        for (int j = 0; j < 16; ++j) {
            float f = krow[chunk + klo + (j & 7) + ((j >> 3) << 4)];
            __bf16 h = (__bf16)f;
            bh[j] = h;
            bl[j] = (__bf16)(f - (float)h);
        }
        // dot ~= ql*kh + qh*kl + qh*kh  (f32 accumulate in the matrix pipe)
        acc = __builtin_amdgcn_wmma_f32_16x16x32_bf16(false, al, false, bh, (short)0, acc, false, false);
        acc = __builtin_amdgcn_wmma_f32_16x16x32_bf16(false, ah, false, bl, (short)0, acc, false, false);
        acc = __builtin_amdgcn_wmma_f32_16x16x32_bf16(false, ah, false, bh, (short)0, acc, false, false);
    }

    // Diagonal extraction: edge i<8 -> lane i, vgpr i; edge i>=8 -> lane i+16, vgpr i-8.
    bool active = (lane < 8) | (lane >= 24);
    int r = ((lane < 8) ? lane : (lane - 24)) & 7;
    float mine = acc[0];
    mine = (r == 1) ? acc[1] : mine;
    mine = (r == 2) ? acc[2] : mine;
    mine = (r == 3) ? acc[3] : mine;
    mine = (r == 4) ? acc[4] : mine;
    mine = (r == 5) ? acc[5] : mine;
    mine = (r == 6) ? acc[6] : mine;
    mine = (r == 7) ? acc[7] : mine;

    if (active) {
        int eo = eBase + (lane & 15);    // equals eBase + ei for active lanes
        if (eo < E) {
            float v = mine + bias[eo];
            logit[(size_t)eo * BATCH + batch] = v;
            // order-preserving uint key: f>=0 -> u|0x80000000 ; f<0 -> ~u
            unsigned u = __float_as_uint(v);
            unsigned key = (u & 0x80000000u) ? ~u : (u | 0x80000000u);
            atomicMax(&smax[(size_t)drow * BATCH + batch], key);
        }
    }
}

// ---------------------------------------------------------------------------
// Kernel 3: exp(logit - segmax[dst]) and atomic segment-sum of denominators.
// Overwrites logit[] with exp values.
// ---------------------------------------------------------------------------
__global__ __launch_bounds__(256) void k_exp(
    const int* __restrict__ dst, const unsigned* __restrict__ smax,
    float* __restrict__ logit, float* __restrict__ denom, int E)
{
    int idx = blockIdx.x * blockDim.x + threadIdx.x;
    if (idx >= E * BATCH) return;
    int e = idx >> 1;
    int b = idx & 1;
    int d = dst[e];
    unsigned key = smax[(size_t)d * BATCH + b];
    unsigned u = (key & 0x80000000u) ? (key & 0x7FFFFFFFu) : ~key;
    float m = __uint_as_float(u);
    float ex = expf(logit[idx] - m);
    logit[idx] = ex;
    atomicAdd(&denom[(size_t)d * BATCH + b], ex);
}

// ---------------------------------------------------------------------------
// Kernel 4: out[b,dst,c] += (exp/denom) * V[b,src,c]   (f32 atomics in L2)
// ---------------------------------------------------------------------------
__global__ __launch_bounds__(256) void k_agg(
    const float* __restrict__ V, const int* __restrict__ src,
    const int* __restrict__ dst, const float* __restrict__ expl,
    const float* __restrict__ denom, float* __restrict__ out,
    int N, int E)
{
    long long idx = (long long)blockIdx.x * blockDim.x + threadIdx.x;
    long long total = (long long)E * BATCH * C_DIM;
    if (idx >= total) return;
    int c = (int)(idx & (C_DIM - 1));
    int b = (int)((idx >> 6) & 1);
    int e = (int)(idx >> 7);
    int d = dst[e];
    int s = src[e];
    float alpha = expl[(size_t)e * BATCH + b] / denom[(size_t)d * BATCH + b];
    float contrib = alpha * V[((size_t)b * N + s) * C_DIM + c];
    atomicAdd(&out[((size_t)b * N + d) * C_DIM + c], contrib);
}

// ---------------------------------------------------------------------------
extern "C" void kernel_launch(void* const* d_in, const int* in_sizes, int n_in,
                              void* d_out, int out_size, void* d_ws, size_t ws_size,
                              hipStream_t stream)
{
    (void)n_in; (void)ws_size;
    const float* Q     = (const float*)d_in[0];
    const float* K     = (const float*)d_in[1];
    const float* V     = (const float*)d_in[2];
    const float* efeat = (const float*)d_in[3];
    const float* W1    = (const float*)d_in[4];
    const float* b1    = (const float*)d_in[5];
    const float* W2    = (const float*)d_in[6];
    const float* b2    = (const float*)d_in[7];
    const int*   src   = (const int*)d_in[8];
    const int*   dst   = (const int*)d_in[9];
    float*       out   = (float*)d_out;

    const int E = in_sizes[8];                       // edges
    const int N = in_sizes[0] / (BATCH * C_DIM);     // nodes

    // workspace layout (floats)
    float*    bias  = (float*)d_ws;                  // E
    float*    logit = bias + E;                      // E*BATCH (logits, then exp)
    unsigned* smax  = (unsigned*)(logit + (size_t)E * BATCH);  // N*BATCH keys
    float*    denom = (float*)(smax + (size_t)N * BATCH);      // N*BATCH

    // per-call init (atomics accumulate -> must zero every launch)
    hipMemsetAsync(smax,  0, (size_t)N * BATCH * sizeof(unsigned), stream);
    hipMemsetAsync(denom, 0, (size_t)N * BATCH * sizeof(float),    stream);
    hipMemsetAsync(out,   0, (size_t)out_size * sizeof(float),     stream);

    // 1) edge bias MLP
    k_bias<<<(E + 255) / 256, 256, 0, stream>>>(efeat, W1, b1, W2, b2, bias, E);

    // 2) WMMA logits + segment max (16 edges per wave, 2 batches)
    {
        int nGroups = (E + 15) >> 4;
        long long waves   = (long long)nGroups * BATCH;
        long long threads = waves * 32;
        int blocks = (int)((threads + 255) / 256);
        k_logits<<<blocks, 256, 0, stream>>>(Q, K, src, dst, bias, logit, smax, N, E);
    }

    // 3) exp + segment sum
    k_exp<<<(E * BATCH + 255) / 256, 256, 0, stream>>>(dst, smax, logit, denom, E);

    // 4) weighted scatter-aggregate
    {
        long long total = (long long)E * BATCH * C_DIM;
        int blocks = (int)((total + 255) / 256);
        k_agg<<<blocks, 256, 0, stream>>>(V, src, dst, logit, denom, out, N, E);
    }
}
